// MultiheadAttention_74594991997239
// MI455X (gfx1250) — matrix operand order repaired and checked by reading, not compiled, not run on hardware
//
#include <hip/hip_runtime.h>
#include <hip/hip_bf16.h>

// ---------------- types / WMMA helper ----------------
typedef __attribute__((ext_vector_type(16))) __bf16 bf16x16;
typedef __attribute__((ext_vector_type(8)))  float  f32x8;
typedef unsigned int __attribute__((ext_vector_type(4))) u32x4;
typedef int  __attribute__((ext_vector_type(4))) i32x4v;
typedef int  __attribute__((ext_vector_type(8))) i32x8v;

union FragB16 { bf16x16 f; uint4 q[2]; };

__device__ __forceinline__ f32x8 wmma_bf16(bf16x16 a, bf16x16 b, f32x8 c) {
  return __builtin_amdgcn_wmma_f32_16x16x32_bf16(false, a, false, b, (short)0, c,
                                                 false, false);
}

#define S_LEN 2048
#define DMODEL 1024
#define NHEAD 16
#define HDIM 64
#define BATCH 2
#define MTOT (BATCH * S_LEN)   // 4096

// 1/sqrt(64) * log2(e): scores end up in log2 domain -> exp2 softmax
#define QSCALE 0.18033688011112042f

// ---------------- Tensor Data Mover: 2-D tile -> LDS ----------------
// D# per CDNA5 ISA. tile_w elems (2B), tile_h rows.
// pad codes: interval 2^(c+1) DWORDs, amount (c+1... code c -> c+1) DWORDs.
__device__ __forceinline__ unsigned lds_off(const void* p) {
  return (unsigned)(uintptr_t)p;   // low 32 bits of generic ptr = LDS byte offset
}

__device__ __forceinline__ void tdm_load_2d(unsigned ldsoff, const void* gptr,
                                            unsigned tile_w, unsigned tile_h,
                                            unsigned row_stride,
                                            unsigned pad_int, unsigned pad_amt)
{
  unsigned long long ga = (unsigned long long)(uintptr_t)gptr;
  u32x4 g0 = { 1u,                                   // count=1 (valid), user mode
               ldsoff,                               // lds_addr
               (unsigned)ga,                         // global_addr[31:0]
               (unsigned)((ga >> 32) & 0x01FFFFFFu) | 0x80000000u }; // ga[56:32] | type=2
  i32x8v g1 = {
    (int)((1u << 16) | (1u << 20) | (pad_int << 22) | (pad_amt << 25)), // 2B elems, pad_en
    (int)((tile_w & 0xFFFFu) << 16),        // tensor_dim0[15:0]
    (int)((tile_h & 0xFFFFu) << 16),        // tensor_dim0[31:16]=0 | tensor_dim1[15:0]
    (int)((tile_w & 0xFFFFu) << 16),        // tensor_dim1[31:16]=0 | tile_dim0
    (int)(tile_h & 0xFFFFu),                // tile_dim1 | tile_dim2=0
    (int)row_stride,                        // tensor_dim0_stride[31:0]
    0, 0 };
  i32x4v gz = { 0, 0, 0, 0 };
#if __clang_major__ >= 23
  i32x8v gz8 = { 0, 0, 0, 0, 0, 0, 0, 0 };
  __builtin_amdgcn_tensor_load_to_lds(g0, g1, gz, gz, gz8, 0);
#else
  __builtin_amdgcn_tensor_load_to_lds(g0, g1, gz, gz, 0);
#endif
}

// ---------------- fp32 -> bf16 convert (4 elems/thread) ----------------
__global__ void cvt_f32_bf16(const float* __restrict__ in, __bf16* __restrict__ out,
                             int n4) {
  int i = blockIdx.x * blockDim.x + threadIdx.x;
  if (i < n4) {
    float4 f = ((const float4*)in)[i];
    __bf16 t[4] = { (__bf16)f.x, (__bf16)f.y, (__bf16)f.z, (__bf16)f.w };
    ((uint2*)out)[i] = *(uint2*)t;
  }
}

// ---------------- GEMM: C = X @ W^T + b, TDM-fed, per-mode epilogue ----------------
// mode 0: Q (RoPE + QSCALE, bf16 [B,H,S,hd]); 1: K (RoPE); 2: V; 3: fp32 [M,N].
// Block tile 128x128, 8 waves (4M x 2N), K-step 64 -> 16 WMMA / wave / barrier pair.
__global__ void __launch_bounds__(256)
gemm_qkvo(const __bf16* __restrict__ X, const __bf16* __restrict__ W,
          const float* __restrict__ bias,
          __bf16* __restrict__ outH, float* __restrict__ outF, int mode)
{
  __shared__ __bf16 As[2][128][72];  // 64 elems + 8 pad: TDM pads 128B rows by 16B
  __shared__ __bf16 Bs[2][128][72];

  const int tid   = threadIdx.x;
  const int wave  = tid >> 5, lane = tid & 31;
  const int lo16  = lane & 15, hi = (lane >> 4) & 1;
  const int waveM = wave >> 1, waveN = wave & 1;
  const int m0    = blockIdx.y * 128, n0 = blockIdx.x * 128;

  f32x8 acc[2][4];
#pragma unroll
  for (int mi = 0; mi < 2; ++mi)
#pragma unroll
    for (int j = 0; j < 4; ++j)
      acc[mi][j] = (f32x8){0.f,0.f,0.f,0.f,0.f,0.f,0.f,0.f};

  if (wave == 0) {   // TDM: one wave drives the DMA
    tdm_load_2d(lds_off(&As[0][0][0]), X + (size_t)m0 * DMODEL, 64, 128, DMODEL, 4, 3);
    tdm_load_2d(lds_off(&Bs[0][0][0]), W + (size_t)n0 * DMODEL, 64, 128, DMODEL, 4, 3);
  }

  for (int k0 = 0; k0 < DMODEL; k0 += 64) {
    const int cur = (k0 >> 6) & 1, nxt = cur ^ 1;
    __syncthreads();                       // buf[nxt] free (prev compute done)
    if (wave == 0) {
      if (k0 + 64 < DMODEL) {
        tdm_load_2d(lds_off(&As[nxt][0][0]),
                    X + (size_t)m0 * DMODEL + (k0 + 64), 64, 128, DMODEL, 4, 3);
        tdm_load_2d(lds_off(&Bs[nxt][0][0]),
                    W + (size_t)n0 * DMODEL + (k0 + 64), 64, 128, DMODEL, 4, 3);
        __builtin_amdgcn_s_wait_tensorcnt(2);  // in-order: current tiles complete
      } else {
        __builtin_amdgcn_s_wait_tensorcnt(0);
      }
    }
    __syncthreads();                       // buf[cur] visible to all waves

    const int koff = hi ? 8 : 0, bk = hi ? 8 : 0;
    FragB16 a[2][2];                       // [kk][mi]
#pragma unroll
    for (int kk = 0; kk < 2; ++kk)
#pragma unroll
      for (int mi = 0; mi < 2; ++mi) {
        const int row = waveM * 32 + mi * 16 + lo16;
        a[kk][mi].q[0] = *(const uint4*)&As[cur][row][kk * 32 + koff];
        a[kk][mi].q[1] = *(const uint4*)&As[cur][row][kk * 32 + koff + 16];
      }
#pragma unroll
    for (int j = 0; j < 4; ++j) {
      const int row = waveN * 64 + j * 16 + lo16;
#pragma unroll
      for (int kk = 0; kk < 2; ++kk) {
        FragB16 b;
        b.q[0] = *(const uint4*)&Bs[cur][row][kk * 32 + bk];
        b.q[1] = *(const uint4*)&Bs[cur][row][kk * 32 + bk + 16];
#pragma unroll
        for (int mi = 0; mi < 2; ++mi)
          acc[mi][j] = wmma_bf16(a[kk][mi].f, b.f, acc[mi][j]);
      }
    }
  }

  // ------- epilogue -------
#pragma unroll
  for (int mi = 0; mi < 2; ++mi) {
#pragma unroll
    for (int j = 0; j < 4; ++j) {
      const int n  = n0 + waveN * 64 + j * 16 + lo16;
      const float bv = bias[n];
#pragma unroll
      for (int r = 0; r < 8; ++r) {
        const int m = m0 + waveM * 32 + mi * 16 + r + 8 * hi;
        float val = acc[mi][j][r] + bv;
        if (mode <= 1) {
          const int s   = m & (S_LEN - 1);
          const int b_  = m >> 11;
          const int hh  = (n >> 6) & (NHEAD - 1);
          const int dgl = n & (HDIM - 1);
          const float inv = __expf(-((float)(dgl & ~1)) * (9.2103403719761836f / 64.f));
          float sn, cs;
          __sincosf((float)s * inv, &sn, &cs);
          const float partner = __shfl_xor(val, 1, 32);   // RoPE pair = adjacent lane
          float rot = (dgl & 1) ? (val * cs + partner * sn)
                                : (val * cs - partner * sn);
          if (mode == 0) rot *= QSCALE;   // 1/sqrt(hd) * log2(e) folded into Q
          outH[((size_t)((b_ * NHEAD + hh) * S_LEN + s) << 6) + dgl] = (__bf16)rot;
        } else if (mode == 2) {
          const int s   = m & (S_LEN - 1);
          const int b_  = m >> 11;
          const int hh  = (n >> 6) & (NHEAD - 1);
          const int dgl = n & (HDIM - 1);
          outH[((size_t)((b_ * NHEAD + hh) * S_LEN + s) << 6) + dgl] = (__bf16)val;
        } else {
          outF[(size_t)m * DMODEL + n] = val;
        }
      }
    }
  }
}

// ---------------- flash attention ----------------
// grid (S/128, H, B), 8 waves x 16 queries. 64-key tiles, K via TDM (double-buffered),
// V transposed manually into LDS. 16 WMMA / wave / tile. exp2-domain softmax.
__global__ void __launch_bounds__(256)
flash_attn(const __bf16* __restrict__ q, const __bf16* __restrict__ k,
           const __bf16* __restrict__ v, __bf16* __restrict__ o)
{
  __shared__ __bf16 Ks[2][64][72];     // TDM pads 128B rows by 16B -> 144B stride
  __shared__ __bf16 Vt[64][72];        // transposed [hd][key]
  __shared__ __bf16 Ps[8][16][72];     // per-wave P staging

  const int tid  = threadIdx.x, wave = tid >> 5, lane = tid & 31;
  const int lo16 = lane & 15, hi = (lane >> 4) & 1;
  const int b = blockIdx.z, h = blockIdx.y;
  const int q0 = blockIdx.x * 128 + wave * 16;
  const size_t bh = ((size_t)(b * NHEAD + h)) * S_LEN * HDIM;

  FragB16 qa[2];
  {
    const __bf16* qp = q + bh + (size_t)(q0 + lo16) * HDIM;
    const int koff = hi ? 8 : 0;
#pragma unroll
    for (int kk = 0; kk < 2; ++kk) {
      qa[kk].q[0] = *(const uint4*)(qp + kk * 32 + koff);
      qa[kk].q[1] = *(const uint4*)(qp + kk * 32 + koff + 16);
    }
  }

  f32x8 Oacc[4];
#pragma unroll
  for (int j = 0; j < 4; ++j) Oacc[j] = (f32x8){0.f,0.f,0.f,0.f,0.f,0.f,0.f,0.f};
  float mrow[8], lrow[8];
#pragma unroll
  for (int r = 0; r < 8; ++r) { mrow[r] = -3.0e38f; lrow[r] = 0.f; }

  if (wave == 0)
    tdm_load_2d(lds_off(&Ks[0][0][0]), k + bh, 64, 64, 64, 4, 3);

  for (int kt = 0; kt < S_LEN; kt += 64) {
    const int cur = (kt >> 6) & 1, nxt = cur ^ 1;
    __syncthreads();                    // prev tile reads done
    if (wave == 0) {
      if (kt + 64 < S_LEN) {
        tdm_load_2d(lds_off(&Ks[nxt][0][0]), k + bh + (size_t)(kt + 64) * HDIM,
                    64, 64, 64, 4, 3);
        __builtin_amdgcn_s_wait_tensorcnt(1);
      } else {
        __builtin_amdgcn_s_wait_tensorcnt(0);
      }
    }
    // V tile -> Vt (transpose on store), all 256 threads, 2 x b128 each
#pragma unroll
    for (int i = 0; i < 2; ++i) {
      const int c = tid + i * 256, key = c >> 3, co = (c & 7) * 8;
      if (kt + 64 < S_LEN)
        __builtin_prefetch(v + bh + (size_t)(kt + 64 + key) * HDIM + co, 0, 0);
      uint4 d = *(const uint4*)(v + bh + (size_t)(kt + key) * HDIM + co);
      __bf16 tmp[8];
      *(uint4*)tmp = d;
#pragma unroll
      for (int jj = 0; jj < 8; ++jj) Vt[co + jj][key] = tmp[jj];
    }
    __syncthreads();                    // K tile (TDM) + Vt ready

    // scores (log2 domain): 16 queries x 64 keys
    f32x8 S[4];
#pragma unroll
    for (int j = 0; j < 4; ++j) S[j] = (f32x8){0.f,0.f,0.f,0.f,0.f,0.f,0.f,0.f};
#pragma unroll
    for (int kk = 0; kk < 2; ++kk) {
      const int bk = kk * 32 + (hi ? 8 : 0);
#pragma unroll
      for (int j = 0; j < 4; ++j) {
        FragB16 kb;
        kb.q[0] = *(const uint4*)&Ks[cur][j * 16 + lo16][bk];
        kb.q[1] = *(const uint4*)&Ks[cur][j * 16 + lo16][bk + 16];
        S[j] = wmma_bf16(qa[kk].f, kb.f, S[j]);
      }
    }

    // online softmax, exp2 (v_exp_f32 is base-2 native)
    float alpha[8];
#pragma unroll
    for (int r = 0; r < 8; ++r) {
      float mx = fmaxf(fmaxf(S[0][r], S[1][r]), fmaxf(S[2][r], S[3][r]));
#pragma unroll
      for (int off = 1; off < 16; off <<= 1) mx = fmaxf(mx, __shfl_xor(mx, off, 32));
      const float mnew = fmaxf(mrow[r], mx);
      const float al   = exp2f(mrow[r] - mnew);
      float p[4], ps = 0.f;
#pragma unroll
      for (int j = 0; j < 4; ++j) { p[j] = exp2f(S[j][r] - mnew); ps += p[j]; }
#pragma unroll
      for (int off = 1; off < 16; off <<= 1) ps += __shfl_xor(ps, off, 32);
      lrow[r] = lrow[r] * al + ps;
      mrow[r] = mnew;
      alpha[r] = al;
#pragma unroll
      for (int j = 0; j < 4; ++j)
        Ps[wave][r + 8 * hi][j * 16 + lo16] = (__bf16)p[j];
    }
#pragma unroll
    for (int j = 0; j < 4; ++j)
#pragma unroll
      for (int r = 0; r < 8; ++r) Oacc[j][r] *= alpha[r];

    // P back as A-fragments (wave-private LDS, DS in-order within wave)
    FragB16 pa[2];
    {
      const int koff = hi ? 8 : 0;
#pragma unroll
      for (int kk2 = 0; kk2 < 2; ++kk2) {
        pa[kk2].q[0] = *(const uint4*)&Ps[wave][lo16][kk2 * 32 + koff];
        pa[kk2].q[1] = *(const uint4*)&Ps[wave][lo16][kk2 * 32 + koff + 16];
      }
    }

    // O += P(16x64) x V(64x64)
#pragma unroll
    for (int kk2 = 0; kk2 < 2; ++kk2) {
      const int vk = kk2 * 32 + (hi ? 8 : 0);
#pragma unroll
      for (int j2 = 0; j2 < 4; ++j2) {
        FragB16 vb;
        vb.q[0] = *(const uint4*)&Vt[j2 * 16 + lo16][vk];
        vb.q[1] = *(const uint4*)&Vt[j2 * 16 + lo16][vk + 16];
        Oacc[j2] = wmma_bf16(pa[kk2].f, vb.f, Oacc[j2]);
      }
    }
  }

  // normalize + store attn output [B,S,H*hd]
#pragma unroll
  for (int j2 = 0; j2 < 4; ++j2) {
#pragma unroll
    for (int r = 0; r < 8; ++r) {
      const int s = q0 + r + 8 * hi;
      const int n = h * HDIM + j2 * 16 + lo16;
      o[((size_t)b * S_LEN + s) * DMODEL + n] = (__bf16)(Oacc[j2][r] / lrow[r]);
    }
  }
}

// ---------------- host launch ----------------
extern "C" void kernel_launch(void* const* d_in, const int* in_sizes, int n_in,
                              void* d_out, int out_size, void* d_ws, size_t ws_size,
                              hipStream_t stream) {
  const float* x  = (const float*)d_in[0];
  const float* Wq = (const float*)d_in[1];
  const float* bq = (const float*)d_in[2];
  const float* Wk = (const float*)d_in[3];
  const float* bk = (const float*)d_in[4];
  const float* Wv = (const float*)d_in[5];
  const float* bv = (const float*)d_in[6];
  const float* Wo = (const float*)d_in[7];
  const float* bo = (const float*)d_in[8];

  const size_t NX = (size_t)MTOT * DMODEL;                 // 4M
  const size_t NW = (size_t)DMODEL * DMODEL;               // 1M
  const size_t NH = (size_t)BATCH * NHEAD * S_LEN * HDIM;  // 4M

  __bf16* xbf = (__bf16*)d_ws;
  __bf16* wqb = xbf + NX;
  __bf16* wkb = wqb + NW;
  __bf16* wvb = wkb + NW;
  __bf16* wob = wvb + NW;
  __bf16* qb  = wob + NW;
  __bf16* kb  = qb + NH;
  __bf16* vb  = kb + NH;
  __bf16* ob  = vb + NH;   // total 24M bf16 = 48 MB

  cvt_f32_bf16<<<(int)((NX / 4 + 255) / 256), 256, 0, stream>>>(x,  xbf, (int)(NX / 4));
  cvt_f32_bf16<<<(int)((NW / 4 + 255) / 256), 256, 0, stream>>>(Wq, wqb, (int)(NW / 4));
  cvt_f32_bf16<<<(int)((NW / 4 + 255) / 256), 256, 0, stream>>>(Wk, wkb, (int)(NW / 4));
  cvt_f32_bf16<<<(int)((NW / 4 + 255) / 256), 256, 0, stream>>>(Wv, wvb, (int)(NW / 4));
  cvt_f32_bf16<<<(int)((NW / 4 + 255) / 256), 256, 0, stream>>>(Wo, wob, (int)(NW / 4));

  dim3 gG(DMODEL / 128, MTOT / 128);   // (8, 32)
  gemm_qkvo<<<gG, 256, 0, stream>>>(xbf, wqb, bq, qb, nullptr, 0);
  gemm_qkvo<<<gG, 256, 0, stream>>>(xbf, wkb, bk, kb, nullptr, 1);
  gemm_qkvo<<<gG, 256, 0, stream>>>(xbf, wvb, bv, vb, nullptr, 2);

  flash_attn<<<dim3(S_LEN / 128, NHEAD, BATCH), 256, 0, stream>>>(qb, kb, vb, ob);

  gemm_qkvo<<<gG, 256, 0, stream>>>(ob, wob, bo, nullptr, (float*)d_out, 3);
}
